// IntraAttention_67886253080733
// MI455X (gfx1250) — compile-verified, matrix operand order checked by reading, not run on hardware
//
#include <hip/hip_runtime.h>
#include <hip/hip_bf16.h>

typedef __attribute__((ext_vector_type(16))) _Float16 v16h;
typedef __attribute__((ext_vector_type(8)))  _Float16 v8h;
typedef __attribute__((ext_vector_type(8)))  float    v8f;
typedef __attribute__((ext_vector_type(4)))  int      v4i;

#define DIM    512
#define HEADS  8
#define DHEAD  64
#define INNER  512
#define SEQ    2048
#define BATCH  4
#define ROWS   (BATCH * SEQ)     /* 8192 */
#define QKSCALE 0.125f           /* 64^-0.5 */

#if __has_builtin(__builtin_amdgcn_global_load_async_to_lds_b128)
#define HAVE_ASYNC_LDS 1
#else
#define HAVE_ASYNC_LDS 0
#endif

// ---------------------------------------------------------------------------
// WMMA fragment loader (wave32, 16x16x32 f16).
// Lane L holds row/col (L&15); K chunks {halfk..halfk+7} and
// {halfk+16..halfk+23}, halfk=(L>>4)*8 => two contiguous b128 loads when the
// operand is K-contiguous per row (global or LDS).
// ---------------------------------------------------------------------------
__device__ __forceinline__ v16h frag_ld(const _Float16* base, int ld) {
  const int lane  = threadIdx.x & 31;
  const int row   = lane & 15;
  const int halfk = (lane >> 4) * 8;
  const _Float16* p = base + row * ld + halfk;
  v8h lo = *(const v8h*)p;
  v8h hi = *(const v8h*)(p + 16);
  return __builtin_shufflevector(lo, hi, 0, 1, 2, 3, 4, 5, 6, 7,
                                         8, 9, 10, 11, 12, 13, 14, 15);
}

__device__ __forceinline__ v8f wmma32(v16h a, v16h b, v8f c) {
  return __builtin_amdgcn_wmma_f32_16x16x32_f16(
      false, a, false, b, (short)0, c, false, false);
}

// 16-byte global -> LDS copy; async (ASYNCcnt / no VGPR round-trip) if the
// toolchain exposes the gfx1250 builtin, else a plain staged copy.
__device__ __forceinline__ void cp16_g2lds(const _Float16* g, _Float16* l) {
#if HAVE_ASYNC_LDS
  __builtin_amdgcn_global_load_async_to_lds_b128(
      (__attribute__((address_space(1))) v4i*)g,
      (__attribute__((address_space(3))) v4i*)l, 0, 0);
#else
  *(float4*)l = *(const float4*)g;
#endif
}

__device__ __forceinline__ void wait_async_lds() {
#if HAVE_ASYNC_LDS
#if __has_builtin(__builtin_amdgcn_s_wait_asynccnt)
  __builtin_amdgcn_s_wait_asynccnt(0);
#else
  asm volatile("s_wait_asynccnt 0x0" ::: "memory");
#endif
#endif
}

// ---------------------------------------------------------------------------
// Kernel 0: fp32 -> f16 TRANSPOSED weight conversion: dst[n][k] = src[k][n].
// ---------------------------------------------------------------------------
__global__ __launch_bounds__(256)
void transpose_cvt_kernel(const float* __restrict__ src, _Float16* __restrict__ dst,
                          int K, int N) {
  __shared__ float tile[32][33];
  const int kb = blockIdx.y * 32, nb = blockIdx.x * 32;
  const int tx = threadIdx.x & 31, ty = threadIdx.x >> 5;
#pragma unroll
  for (int r = ty; r < 32; r += 8)
    tile[r][tx] = src[(size_t)(kb + r) * N + nb + tx];
  __syncthreads();
#pragma unroll
  for (int r = ty; r < 32; r += 8)
    dst[(size_t)(nb + r) * K + kb + tx] = (_Float16)tile[tx][r];
}

// ---------------------------------------------------------------------------
// Kernel 1: LayerNorm over DIM=512, emit f16 activations. 128 thr/row.
// ---------------------------------------------------------------------------
__global__ __launch_bounds__(128)
void ln_kernel(const float* __restrict__ x, const float* __restrict__ gamma,
               const float* __restrict__ beta, _Float16* __restrict__ h) {
  const int row = blockIdx.x;
  const float4* xr = (const float4*)(x + (size_t)row * DIM);
  float4 v = xr[threadIdx.x];
  float s = v.x + v.y + v.z + v.w;
  float q = v.x * v.x + v.y * v.y + v.z * v.z + v.w * v.w;
#pragma unroll
  for (int m = 16; m >= 1; m >>= 1) {
    s += __shfl_xor(s, m, 32);
    q += __shfl_xor(q, m, 32);
  }
  __shared__ float ss[4], qq[4];
  if ((threadIdx.x & 31) == 0) {
    ss[threadIdx.x >> 5] = s;
    qq[threadIdx.x >> 5] = q;
  }
  __syncthreads();
  s = ss[0] + ss[1] + ss[2] + ss[3];
  q = qq[0] + qq[1] + qq[2] + qq[3];
  const float mu   = s * (1.0f / DIM);
  const float var  = q * (1.0f / DIM) - mu * mu;
  const float rstd = rsqrtf(var + 1e-5f);
  _Float16* hr = h + (size_t)row * DIM;
  const int c = threadIdx.x * 4;
  hr[c + 0] = (_Float16)((v.x - mu) * rstd * gamma[c + 0] + beta[c + 0]);
  hr[c + 1] = (_Float16)((v.y - mu) * rstd * gamma[c + 1] + beta[c + 1]);
  hr[c + 2] = (_Float16)((v.z - mu) * rstd * gamma[c + 2] + beta[c + 2]);
  hr[c + 3] = (_Float16)((v.w - mu) * rstd * gamma[c + 3] + beta[c + 3]);
}

// ---------------------------------------------------------------------------
// Kernel 2: QKV GEMM. B given transposed (Bt[n][k]); fragments load straight
// from global (weights L2-resident). 8 waves x (32x32) = 128x64 block tile.
// Epilogue: Q/K -> [B,H,N,64] (Q scaled), V -> [B,H,64,N] via packed b128.
// ---------------------------------------------------------------------------
__global__ __launch_bounds__(256)
void gemm_qkv_kernel(const _Float16* __restrict__ A, const _Float16* __restrict__ Bt,
                     _Float16* __restrict__ q16, _Float16* __restrict__ k16,
                     _Float16* __restrict__ v16) {
  const int wid = threadIdx.x >> 5;
  const int wm = blockIdx.y * 128 + (wid & 3) * 32;
  const int wn = blockIdx.x * 64 + (wid >> 2) * 32;
  const _Float16* Ar0 = A + (size_t)wm * DIM;
  const _Float16* Ar1 = A + (size_t)(wm + 16) * DIM;
  const _Float16* Br0 = Bt + (size_t)wn * DIM;
  const _Float16* Br1 = Bt + (size_t)(wn + 16) * DIM;
  v8f acc00 = {}, acc01 = {}, acc10 = {}, acc11 = {};
#pragma unroll 4
  for (int kk = 0; kk < DIM; kk += 32) {
    v16h a0 = frag_ld(Ar0 + kk, DIM);
    v16h a1 = frag_ld(Ar1 + kk, DIM);
    v16h b0 = frag_ld(Br0 + kk, DIM);
    v16h b1 = frag_ld(Br1 + kk, DIM);
    acc00 = wmma32(a0, b0, acc00);
    acc01 = wmma32(a0, b1, acc01);
    acc10 = wmma32(a1, b0, acc10);
    acc11 = wmma32(a1, b1, acc11);
  }
  const int lane  = threadIdx.x & 31;
  const int colL  = lane & 15;
  const int rbase = (lane >> 4) * 8;
  const int mtx   = (blockIdx.x * 64) >> 9;   // block-uniform: 0=Q 1=K 2=V
#pragma unroll
  for (int rt = 0; rt < 2; ++rt) {
#pragma unroll
    for (int ct = 0; ct < 2; ++ct) {
      const v8f acc = (rt == 0) ? (ct == 0 ? acc00 : acc01)
                                : (ct == 0 ? acc10 : acc11);
      const int m0  = wm + rt * 16 + rbase;         // 8 contiguous rows/lane
      const int c   = wn + ct * 16 + colL;
      const int bb  = m0 >> 11;
      const int nn0 = m0 & 2047;
      const int hh  = (c & 511) >> 6;
      const int dd  = c & 63;
      const size_t head = (size_t)(bb * HEADS + hh);
      if (mtx == 2) {            // V dim-major: 8 rows are contiguous -> b128
        v8h pk;
#pragma unroll
        for (int i = 0; i < 8; ++i) pk[i] = (_Float16)acc[i];
        *(v8h*)&v16[(head * DHEAD + dd) * SEQ + nn0] = pk;
      } else if (mtx == 1) {
        _Float16* p = k16 + (head * SEQ + nn0) * DHEAD + dd;
#pragma unroll
        for (int i = 0; i < 8; ++i) p[(size_t)i * DHEAD] = (_Float16)acc[i];
      } else {
        _Float16* p = q16 + (head * SEQ + nn0) * DHEAD + dd;
#pragma unroll
        for (int i = 0; i < 8; ++i)
          p[(size_t)i * DHEAD] = (_Float16)(acc[i] * QKSCALE);
      }
    }
  }
}

// ---------------------------------------------------------------------------
// Kernel 3: flash attention. Block = (b, h, 128 query rows); 8 waves own a
// 16-row strip. Double-buffered K/V tiles staged via async global->LDS loads
// (ASYNCcnt) so the next block's DMA overlaps this block's WMMA + softmax.
// ---------------------------------------------------------------------------
__global__ __launch_bounds__(256)
void attn_kernel(const _Float16* __restrict__ q16, const _Float16* __restrict__ k16,
                 const _Float16* __restrict__ vt16, _Float16* __restrict__ o16) {
  __shared__ __align__(16) _Float16 Ks[2][64][64];    // [buf][key][d]
  __shared__ __align__(16) _Float16 Vt[2][64][64];    // [buf][d][key]
  __shared__ __align__(16) _Float16 Ps[8][16][64];    // per-wave [row][key]
  const int t    = threadIdx.x;
  const int wid  = t >> 5;
  const int lane = t & 31;
  const int colL  = lane & 15;
  const int rbase = (lane >> 4) * 8;
  const int qb = blockIdx.x, h = blockIdx.y, b = blockIdx.z;
  const int qn0 = qb * 128 + wid * 16;
  const size_t headOff = ((size_t)(b * HEADS + h)) * SEQ * DHEAD;

  const _Float16* qptr = q16 + headOff + (size_t)qn0 * DHEAD;
  const v16h qa0 = frag_ld(qptr, DHEAD);        // d = 0..31
  const v16h qa1 = frag_ld(qptr + 32, DHEAD);   // d = 32..63

  v8f O0 = {}, O1 = {}, O2 = {}, O3 = {};
  float m8[8], l8[8];
#pragma unroll
  for (int i = 0; i < 8; ++i) { m8[i] = -3.0e38f; l8[i] = 0.0f; }

  const int lr = t >> 2;          // 0..63
  const int lc = (t & 3) * 16;    // 0,16,32,48

  auto issue_tiles = [&](int jj, int buf) {
    const _Float16* kg = k16 + headOff + (size_t)(jj + lr) * DHEAD + lc;
    const _Float16* vg = vt16 + headOff + (size_t)lr * SEQ + jj + lc;
    cp16_g2lds(kg,     &Ks[buf][lr][lc]);
    cp16_g2lds(kg + 8, &Ks[buf][lr][lc + 8]);
    cp16_g2lds(vg,     &Vt[buf][lr][lc]);
    cp16_g2lds(vg + 8, &Vt[buf][lr][lc + 8]);
  };

  issue_tiles(0, 0);

  for (int j = 0; j < SEQ; j += 64) {
    const int cur = (j >> 6) & 1;
    wait_async_lds();          // this wave's staging for `cur` has landed
    __syncthreads();           // every wave's staging has landed
    if (j + 64 < SEQ) issue_tiles(j + 64, cur ^ 1);   // overlap next DMA

    // S = Q K^T (keys play the N role; Ks rows are d-contiguous).
    v8f S[4];
#pragma unroll
    for (int tt = 0; tt < 4; ++tt) {
      v8f s = {};
      s = wmma32(qa0, frag_ld(&Ks[cur][tt * 16][0], 64), s);
      s = wmma32(qa1, frag_ld(&Ks[cur][tt * 16][32], 64), s);
      S[tt] = s;
    }

    // Online softmax: row max over 64 keys.
    float tmax[8];
#pragma unroll
    for (int i = 0; i < 8; ++i)
      tmax[i] = fmaxf(fmaxf(S[0][i], S[1][i]), fmaxf(S[2][i], S[3][i]));
#pragma unroll
    for (int mk = 1; mk <= 8; mk <<= 1)
#pragma unroll
      for (int i = 0; i < 8; ++i)
        tmax[i] = fmaxf(tmax[i], __shfl_xor(tmax[i], mk, 32));

    float rsum[8];
#pragma unroll
    for (int i = 0; i < 8; ++i) {
      const float mn   = fmaxf(m8[i], tmax[i]);
      const float corr = __expf(m8[i] - mn);
      m8[i] = mn;
      l8[i] *= corr;
      O0[i] *= corr; O1[i] *= corr; O2[i] *= corr; O3[i] *= corr;
      rsum[i] = 0.0f;
    }
#pragma unroll
    for (int tt = 0; tt < 4; ++tt)
#pragma unroll
      for (int i = 0; i < 8; ++i) {
        const float p = __expf(S[tt][i] - m8[i]);
        rsum[i] += p;
        Ps[wid][rbase + i][tt * 16 + colL] = (_Float16)p;
      }
#pragma unroll
    for (int mk = 1; mk <= 8; mk <<= 1)
#pragma unroll
      for (int i = 0; i < 8; ++i) rsum[i] += __shfl_xor(rsum[i], mk, 32);
#pragma unroll
    for (int i = 0; i < 8; ++i) l8[i] += rsum[i];

    __syncthreads();   // publish P (cross-lane) before re-fragmenting

    const v16h pa0 = frag_ld(&Ps[wid][0][0], 64);    // keys 0..31
    const v16h pa1 = frag_ld(&Ps[wid][0][32], 64);   // keys 32..63
    O0 = wmma32(pa0, frag_ld(&Vt[cur][0][0],  64), O0);
    O0 = wmma32(pa1, frag_ld(&Vt[cur][0][32], 64), O0);
    O1 = wmma32(pa0, frag_ld(&Vt[cur][16][0], 64), O1);
    O1 = wmma32(pa1, frag_ld(&Vt[cur][16][32],64), O1);
    O2 = wmma32(pa0, frag_ld(&Vt[cur][32][0], 64), O2);
    O2 = wmma32(pa1, frag_ld(&Vt[cur][32][32],64), O2);
    O3 = wmma32(pa0, frag_ld(&Vt[cur][48][0], 64), O3);
    O3 = wmma32(pa1, frag_ld(&Vt[cur][48][32],64), O3);
    __syncthreads();   // done reading cur buffer (rewritten at j+2)
  }

  // Normalize and store attention output as [B*N, H*64] f16 (GEMM-ready).
  _Float16* op = o16 + (size_t)(b * SEQ + qn0 + rbase) * INNER + h * DHEAD + colL;
#pragma unroll
  for (int i = 0; i < 8; ++i) {
    const float inv = 1.0f / l8[i];
    op[(size_t)i * INNER +  0] = (_Float16)(O0[i] * inv);
    op[(size_t)i * INNER + 16] = (_Float16)(O1[i] * inv);
    op[(size_t)i * INNER + 32] = (_Float16)(O2[i] * inv);
    op[(size_t)i * INNER + 48] = (_Float16)(O3[i] * inv);
  }
}

// ---------------------------------------------------------------------------
// Kernel 4: output projection, Wout transposed (Bt[n][k]); fp32 -> d_out.
// ---------------------------------------------------------------------------
__global__ __launch_bounds__(256)
void gemm_out_kernel(const _Float16* __restrict__ A, const _Float16* __restrict__ Bt,
                     float* __restrict__ C) {
  const int wid = threadIdx.x >> 5;
  const int wm = blockIdx.y * 128 + (wid & 3) * 32;
  const int wn = blockIdx.x * 64 + (wid >> 2) * 32;
  const _Float16* Ar0 = A + (size_t)wm * INNER;
  const _Float16* Ar1 = A + (size_t)(wm + 16) * INNER;
  const _Float16* Br0 = Bt + (size_t)wn * INNER;
  const _Float16* Br1 = Bt + (size_t)(wn + 16) * INNER;
  v8f acc00 = {}, acc01 = {}, acc10 = {}, acc11 = {};
#pragma unroll 4
  for (int kk = 0; kk < INNER; kk += 32) {
    v16h a0 = frag_ld(Ar0 + kk, INNER);
    v16h a1 = frag_ld(Ar1 + kk, INNER);
    v16h b0 = frag_ld(Br0 + kk, INNER);
    v16h b1 = frag_ld(Br1 + kk, INNER);
    acc00 = wmma32(a0, b0, acc00);
    acc01 = wmma32(a0, b1, acc01);
    acc10 = wmma32(a1, b0, acc10);
    acc11 = wmma32(a1, b1, acc11);
  }
  const int lane  = threadIdx.x & 31;
  const int colL  = lane & 15;
  const int rbase = (lane >> 4) * 8;
#pragma unroll
  for (int rt = 0; rt < 2; ++rt) {
#pragma unroll
    for (int ct = 0; ct < 2; ++ct) {
      const v8f acc = (rt == 0) ? (ct == 0 ? acc00 : acc01)
                                : (ct == 0 ? acc10 : acc11);
      float* p = C + (size_t)(wm + rt * 16 + rbase) * DIM + wn + ct * 16 + colL;
#pragma unroll
      for (int i = 0; i < 8; ++i) p[(size_t)i * DIM] = acc[i];
    }
  }
}

// ---------------------------------------------------------------------------
// Host-side launcher
// ---------------------------------------------------------------------------
extern "C" void kernel_launch(void* const* d_in, const int* in_sizes, int n_in,
                              void* d_out, int out_size, void* d_ws, size_t ws_size,
                              hipStream_t stream) {
  (void)in_sizes; (void)n_in; (void)out_size; (void)ws_size;
  const float* x     = (const float*)d_in[0];
  const float* gamma = (const float*)d_in[1];
  const float* beta  = (const float*)d_in[2];
  const float* wqkv  = (const float*)d_in[3];
  const float* wout  = (const float*)d_in[4];
  float* out = (float*)d_out;

  char* ws = (char*)d_ws;
  size_t off = 0;
  auto carve = [&](size_t bytes) -> void* {
    void* p = ws + off;
    off += (bytes + 255) & ~(size_t)255;
    return p;
  };
  _Float16* h16   = (_Float16*)carve((size_t)ROWS * DIM * 2);
  _Float16* wqkvT = (_Float16*)carve((size_t)(3 * INNER) * DIM * 2);  // [1536][512]
  _Float16* woutT = (_Float16*)carve((size_t)DIM * INNER * 2);        // [512][512]
  _Float16* q16   = (_Float16*)carve((size_t)ROWS * INNER * 2);
  _Float16* k16   = (_Float16*)carve((size_t)ROWS * INNER * 2);
  _Float16* v16   = (_Float16*)carve((size_t)ROWS * INNER * 2);       // dim-major
  _Float16* a16   = (_Float16*)carve((size_t)ROWS * INNER * 2);

  transpose_cvt_kernel<<<dim3((3 * INNER) / 32, DIM / 32), 256, 0, stream>>>(
      wqkv, wqkvT, DIM, 3 * INNER);
  transpose_cvt_kernel<<<dim3(DIM / 32, INNER / 32), 256, 0, stream>>>(
      wout, woutT, INNER, DIM);
  ln_kernel<<<ROWS, 128, 0, stream>>>(x, gamma, beta, h16);
  gemm_qkv_kernel<<<dim3((3 * INNER) / 64, ROWS / 128), 256, 0, stream>>>(
      h16, wqkvT, q16, k16, v16);
  attn_kernel<<<dim3(SEQ / 128, HEADS, BATCH), 256, 0, stream>>>(
      q16, k16, v16, a16);
  gemm_out_kernel<<<dim3(DIM / 64, ROWS / 128), 256, 0, stream>>>(
      a16, woutT, out);
}
